// PointNetMultiSetAbstraction_72524817760259
// MI455X (gfx1250) — matrix operand order repaired and checked by reading, not compiled, not run on hardware
//
#include <hip/hip_runtime.h>
#include <hip/hip_fp16.h>

typedef __attribute__((ext_vector_type(16))) _Float16 v16h;
typedef __attribute__((ext_vector_type(8)))  _Float16 v8h;
typedef __attribute__((ext_vector_type(8)))  float    v8f;

#define BATCH  8
#define NPTS   4096
#define KCENT  1024
#define BN_EPS 1e-5f

// ---------------------------------------------------------------- FPS
// One workgroup per batch. 256 threads, 16 pts/thread cached in VGPRs.
__global__ void fps_kernel(const float* __restrict__ xyz, int* __restrict__ fps_idx,
                           float* __restrict__ new_xyz) {
  const int b = blockIdx.x;
  const int t = threadIdx.x;
  const float* xb = xyz + (size_t)b * 3 * NPTS;
  float px[16], py[16], pz[16], dist[16];
#pragma unroll
  for (int i = 0; i < 16; ++i) {
    int n = i * 256 + t;
    px[i] = xb[n]; py[i] = xb[NPTS + n]; pz[i] = xb[2 * NPTS + n];
    dist[i] = 1e10f;
  }
  __shared__ float rv[256];
  __shared__ int   ri[256];
  __shared__ float cen[3];
  __shared__ int   cur;
  if (t == 0) cur = 0;
  __syncthreads();
  for (int it = 0; it < KCENT; ++it) {
    if (t == 0) {
      int c = cur;
      fps_idx[b * KCENT + it] = c;
      float cx = xb[c], cy = xb[NPTS + c], cz = xb[2 * NPTS + c];
      cen[0] = cx; cen[1] = cy; cen[2] = cz;
      float* nx = new_xyz + ((size_t)b * KCENT + it) * 3;
      nx[0] = cx; nx[1] = cy; nx[2] = cz;
    }
    __syncthreads();
    float cx = cen[0], cy = cen[1], cz = cen[2];
    float best = -1.0f; int bi = 0;
#pragma unroll
    for (int i = 0; i < 16; ++i) {
      float dx = px[i] - cx, dy = py[i] - cy, dz = pz[i] - cz;
      float d = dx * dx + dy * dy + dz * dz;
      float nd = fminf(dist[i], d);
      dist[i] = nd;
      int n = i * 256 + t;
      if (nd > best) { best = nd; bi = n; }   // ascending n => first-occurrence on ties
    }
    rv[t] = best; ri[t] = bi;
    __syncthreads();
    for (int s = 128; s > 0; s >>= 1) {
      if (t < s) {
        float ov = rv[t + s]; int oi = ri[t + s];
        if (ov > rv[t] || (ov == rv[t] && oi < ri[t])) { rv[t] = ov; ri[t] = oi; }
      }
      __syncthreads();
    }
    if (t == 0) cur = ri[0];
    __syncthreads();
  }
}

__global__ void newxyz_out_kernel(const float* __restrict__ new_xyz, float* __restrict__ out) {
  int i = blockIdx.x * blockDim.x + threadIdx.x;   // B*3*K, output is (B,3,K)
  if (i >= BATCH * 3 * KCENT) return;
  int k = i % KCENT; int d = (i / KCENT) % 3; int b = i / (3 * KCENT);
  out[i] = new_xyz[((size_t)b * KCENT + k) * 3 + d];
}

// ---------------------------------------------------------------- Ball query
// One wave32 per centroid; ballot+popc ordered compaction of in-radius indices.
__global__ void ball_query_kernel(const float* __restrict__ xyz, const float* __restrict__ new_xyz,
                                  int* __restrict__ gidx, float r2, int ns) {
  int wid  = blockIdx.x * (blockDim.x >> 5) + (threadIdx.x >> 5);
  int lane = threadIdx.x & 31;
  int b = wid / KCENT;
  const float* xb = xyz + (size_t)b * 3 * NPTS;
  const float* c  = new_xyz + (size_t)wid * 3;
  float cx = c[0], cy = c[1], cz = c[2];
  int* grow = gidx + (size_t)wid * ns;
  int cnt = 0;
  for (int base = 0; base < NPTS && cnt < ns; base += 32) {
    int n = base + lane;
    float dx = xb[n] - cx, dy = xb[NPTS + n] - cy, dz = xb[2 * NPTS + n] - cz;
    bool in = (dx * dx + dy * dy + dz * dz) <= r2;
    unsigned mask = (unsigned)__ballot(in);
    int pos = cnt + __popc(mask & ((1u << lane) - 1u));
    if (in && pos < ns) grow[pos] = n;
    cnt += __popc(mask);
  }
  if (cnt == 0) { for (int j = lane; j < ns; j += 32) grow[j] = 0; return; }
  if (cnt < ns) {
    int first = grow[0];
    for (int j = cnt + lane; j < ns; j += 32) grow[j] = first;
  }
}

// ---------------------------------------------------------------- Gather + concat -> f16 A (K padded to 96)
__global__ void group_kernel(const float* __restrict__ features, const float* __restrict__ xyz,
                             const float* __restrict__ new_xyz, const int* __restrict__ gidx,
                             _Float16* __restrict__ A, int ns, int kpad) {
  int row = blockIdx.x;          // M = B*KCENT*ns rows
  int c   = threadIdx.x;         // kpad (=96) threads
  int bk  = row / ns;
  int b   = bk / KCENT;
  int idx = gidx[(size_t)bk * ns + (row % ns)];
  float v = 0.f;
  if (c < 64) {
    v = features[((size_t)b * 64 + c) * NPTS + idx];
  } else if (c < 67) {
    int d = c - 64;
    v = xyz[((size_t)b * 3 + d) * NPTS + idx] - new_xyz[(size_t)bk * 3 + d];
  }
  A[(size_t)row * kpad + c] = (_Float16)v;
}

// Weights -> f16, zero-padded along K, TRANSPOSED to (N x K) row-major so both
// GEMM tiles stage with contiguous 16B chunks and fragments read contiguous K runs.
__global__ void wpadT_kernel(const float* __restrict__ W, _Float16* __restrict__ WpT,
                             int kreal, int kpad, int nout) {
  int i = blockIdx.x * blockDim.x + threadIdx.x;
  if (i >= kpad * nout) return;
  int n = i / kpad, k = i % kpad;
  WpT[i] = (k < kreal) ? (_Float16)W[k * nout + n] : (_Float16)0.f;
}

// ---------------------------------------------------------------- WMMA GEMM with async-to-LDS double buffering
// Block tile 128x64, 8 waves x (16 rows x 64 cols), K stepped by 32.
// Tiles staged via GLOBAL_LOAD_ASYNC_TO_LDS_B128 (ASYNCcnt DMA): each wave issues
// exactly 3 async ops per tile (2 A chunks + 1 B chunk, uniform EXEC), so
// s_wait_asynccnt 3 covers the current tile while the next streams in.
// A fragment (ISA 16-bit A 16x32 table): lane(m,hi) -> row m, K runs {hi*8..+7},{16+hi*8..+7}
// B fragment (same K order as A): lane n -> column n, K runs {hi*8..+7},{16+hi*8..+7}

__device__ __forceinline__ void async_ld16(unsigned lds_byte, const _Float16* gptr) {
  unsigned long long ga = (unsigned long long)(size_t)gptr;
  asm volatile("global_load_async_to_lds_b128 %0, %1, off"
               :: "v"(lds_byte), "v"(ga) : "memory");
}

__global__ __launch_bounds__(256) void gemm_kernel(const _Float16* __restrict__ A,
                                                   const _Float16* __restrict__ WpT,
                                                   const float* __restrict__ bias,
                                                   float* __restrict__ Y,
                                                   int M, int K, int N) {
  // double-buffered tiles; row stride 40 halves (80B, 16B-aligned)
  __shared__ __align__(16) _Float16 As[2 * 128 * 40];
  __shared__ __align__(16) _Float16 Bs[2 * 64 * 40];
  const int tid  = threadIdx.x;
  const int w    = tid >> 5, lane = tid & 31;
  const int m    = lane & 15, hi = lane >> 4;
  const int tileM = blockIdx.x * 128;
  const int tileN = blockIdx.y * 64;

  const unsigned ldsA0 = (unsigned)(size_t)&As[0];   // generic->LDS offset (low 32 bits)
  const unsigned ldsB0 = (unsigned)(size_t)&Bs[0];
  // per-thread chunk coords: 8-half (16B) chunks, 4 per 32-half row
  const int rA = tid >> 2, cA = (tid & 3) * 8;       // A rows 0..63 (+64 for 2nd chunk)
  const int rB = tid >> 2, cB = (tid & 3) * 8;       // B rows 0..63

  v8f acc[4];
#pragma unroll
  for (int nb = 0; nb < 4; ++nb)
#pragma unroll
    for (int r = 0; r < 8; ++r) acc[nb][r] = 0.f;

  const int nk = K >> 5;
  auto issue_tile = [&](int kt, int buf) {
    int k0 = kt << 5;
    unsigned la = ldsA0 + (unsigned)(buf * 10240 + rA * 80 + cA * 2);
    async_ld16(la,                 A + (size_t)(tileM + rA) * K + k0 + cA);
    async_ld16(la + 64u * 80u,     A + (size_t)(tileM + rA + 64) * K + k0 + cA);
    unsigned lb = ldsB0 + (unsigned)(buf * 5120 + rB * 80 + cB * 2);
    async_ld16(lb,                 WpT + (size_t)(tileN + rB) * K + k0 + cB);
  };

  issue_tile(0, 0);
  for (int kt = 0; kt < nk; ++kt) {
    const int buf = kt & 1;
    if (kt + 1 < nk) {
      issue_tile(kt + 1, buf ^ 1);
      asm volatile("s_wait_asynccnt 0x3" ::: "memory");  // current tile landed
    } else {
      asm volatile("s_wait_asynccnt 0x0" ::: "memory");
    }
    __syncthreads();

    const _Float16* Ab = &As[buf * 5120];
    const _Float16* Bb = &Bs[buf * 2560];
    const int abase = (w * 16 + m) * 40;
    v8h alo = *(const v8h*)(&Ab[abase + hi * 8]);
    v8h ahi = *(const v8h*)(&Ab[abase + 16 + hi * 8]);
    v16h a = __builtin_shufflevector(alo, ahi, 0, 1, 2, 3, 4, 5, 6, 7,
                                     8, 9, 10, 11, 12, 13, 14, 15);
#pragma unroll
    for (int nb = 0; nb < 4; ++nb) {
      const int bbase = (nb * 16 + m) * 40 + hi * 8;
      v8h blo = *(const v8h*)(&Bb[bbase]);
      v8h bhi = *(const v8h*)(&Bb[bbase + 16]);
      v16h bb = __builtin_shufflevector(blo, bhi, 0, 1, 2, 3, 4, 5, 6, 7,
                                        8, 9, 10, 11, 12, 13, 14, 15);
      acc[nb] = __builtin_amdgcn_wmma_f32_16x16x32_f16(
          false, a, false, bb, (short)0, acc[nb], false, false);
    }
    __syncthreads();   // all waves done reading buf before it is overwritten
  }
  // C/D layout: element r -> row r + 8*hi, col = lane&15
#pragma unroll
  for (int nb = 0; nb < 4; ++nb) {
    int col  = tileN + nb * 16 + m;
    float bv = bias[col];
#pragma unroll
    for (int r = 0; r < 8; ++r) {
      int row = tileM + w * 16 + r + hi * 8;
      Y[(size_t)row * N + col] = acc[nb][r] + bv;
    }
  }
}

// ---------------------------------------------------------------- BN stats (deterministic 2-stage)
__global__ void bn_stats_kernel(const float* __restrict__ Y, float* __restrict__ P1,
                                float* __restrict__ P2, int C) {
  int t = threadIdx.x;
  size_t r0 = (size_t)blockIdx.x * 256;
  if (t >= C) return;
  float s = 0.f, s2 = 0.f;
  for (int r = 0; r < 256; ++r) {
    float v = Y[(r0 + r) * (size_t)C + t];
    s += v; s2 += v * v;
  }
  P1[(size_t)blockIdx.x * C + t] = s;
  P2[(size_t)blockIdx.x * C + t] = s2;
}

__global__ void bn_finalize_kernel(const float* __restrict__ P1, const float* __restrict__ P2,
                                   const float* __restrict__ gamma, const float* __restrict__ beta,
                                   float* __restrict__ scale, float* __restrict__ shift,
                                   int nblk, int C, float invM) {
  int c = threadIdx.x;
  if (c >= C) return;
  float s = 0.f, s2 = 0.f;
  for (int i = 0; i < nblk; ++i) { s += P1[(size_t)i * C + c]; s2 += P2[(size_t)i * C + c]; }
  float mean = s * invM;
  float var  = s2 * invM - mean * mean;
  float sc   = gamma[c] * rsqrtf(var + BN_EPS);
  scale[c] = sc;
  shift[c] = beta[c] - mean * sc;
}

__global__ void bn_relu_h_kernel(const float* __restrict__ Y, const float* __restrict__ scale,
                                 const float* __restrict__ shift, _Float16* __restrict__ out,
                                 size_t total, int C) {
  size_t i = (size_t)blockIdx.x * blockDim.x + threadIdx.x;
  if (i >= total) return;
  int c = (int)(i % (size_t)C);
  out[i] = (_Float16)fmaxf(Y[i] * scale[c] + shift[c], 0.f);
}

__global__ void bn_relu_f_kernel(float* __restrict__ Y, const float* __restrict__ scale,
                                 const float* __restrict__ shift, size_t total, int C) {
  size_t i = (size_t)blockIdx.x * blockDim.x + threadIdx.x;
  if (i >= total) return;
  int c = (int)(i % (size_t)C);
  Y[i] = fmaxf(Y[i] * scale[c] + shift[c], 0.f);
}

// ---------------------------------------------------------------- Max over ns -> output (B,640,K) slice
__global__ void maxpool_kernel(const float* __restrict__ Yn, float* __restrict__ out,
                               int ns, int C, int coff) {
  int i = blockIdx.x * blockDim.x + threadIdx.x;   // B*K*C
  if (i >= BATCH * KCENT * C) return;
  int c = i % C; int bk = i / C;
  int b = bk / KCENT, k = bk % KCENT;
  float mx = -3.4e38f;
  for (int j = 0; j < ns; ++j) mx = fmaxf(mx, Yn[((size_t)bk * ns + j) * C + c]);
  out[((size_t)b * 640 + coff + c) * KCENT + k] = mx;
}

// ================================================================ launch
extern "C" void kernel_launch(void* const* d_in, const int* in_sizes, int n_in,
                              void* d_out, int out_size, void* d_ws, size_t ws_size,
                              hipStream_t stream) {
  (void)in_sizes; (void)n_in; (void)out_size; (void)ws_size;
  const float* xyz      = (const float*)d_in[0];
  const float* features = (const float*)d_in[1];
  const float *W[3][3], *bias[3][3], *gamma[3][3], *beta[3][3];
  int p = 2;
  for (int s = 0; s < 3; ++s)
    for (int l = 0; l < 3; ++l) {
      W[s][l]     = (const float*)d_in[p++];
      bias[s][l]  = (const float*)d_in[p++];
      gamma[s][l] = (const float*)d_in[p++];
      beta[s][l]  = (const float*)d_in[p++];
    }
  float* out = (float*)d_out;

  char* wsp = (char*)d_ws;
  size_t off = 0;
  auto carve = [&](size_t bytes) -> void* {
    void* q = wsp + off;
    off += bytes; off = (off + 255) & ~(size_t)255;
    return q;
  };
  const size_t Mmax = (size_t)BATCH * KCENT * 64;
  int*      fps_idx = (int*)carve((size_t)BATCH * KCENT * 4);
  float*    nxyz    = (float*)carve((size_t)BATCH * KCENT * 3 * 4);
  int*      gidx    = (int*)carve((size_t)BATCH * KCENT * 64 * 4);
  float*    scale   = (float*)carve(256 * 4);
  float*    shift   = (float*)carve(256 * 4);
  _Float16* WpT     = (_Float16*)carve((size_t)128 * 256 * 2);
  float*    P1      = (float*)carve((Mmax / 256) * 256 * 4);
  float*    P2      = (float*)carve((Mmax / 256) * 256 * 4);
  _Float16* A0      = (_Float16*)carve(Mmax * 128 * 2);
  _Float16* A1      = (_Float16*)carve(Mmax * 128 * 2);
  float*    Y       = (float*)carve(Mmax * 256 * 4);

  fps_kernel<<<BATCH, 256, 0, stream>>>(xyz, fps_idx, nxyz);
  newxyz_out_kernel<<<(BATCH * 3 * KCENT + 255) / 256, 256, 0, stream>>>(nxyz, out);

  const int   NS[3]      = {16, 32, 64};
  const float RAD[3]     = {0.1f, 0.2f, 0.4f};
  const int   DIMS[3][3] = {{64, 64, 128}, {128, 128, 256}, {128, 128, 256}};
  const int   COFF[3]    = {0, 128, 384};

  for (int s = 0; s < 3; ++s) {
    int ns = NS[s];
    int M  = BATCH * KCENT * ns;
    ball_query_kernel<<<(BATCH * KCENT) / 8, 256, 0, stream>>>(xyz, nxyz, gidx,
                                                               RAD[s] * RAD[s], ns);
    group_kernel<<<M, 96, 0, stream>>>(features, xyz, nxyz, gidx, A0, ns, 96);

    _Float16* Ain = A0;
    int kpad = 96, kreal = 67;
    for (int l = 0; l < 3; ++l) {
      int kout = DIMS[s][l];
      wpadT_kernel<<<(kpad * kout + 255) / 256, 256, 0, stream>>>(W[s][l], WpT, kreal, kpad, kout);
      dim3 gg(M / 128, kout / 64);
      gemm_kernel<<<gg, 256, 0, stream>>>(Ain, WpT, bias[s][l], Y, M, kpad, kout);
      int nblk = M / 256;
      bn_stats_kernel<<<nblk, 256, 0, stream>>>(Y, P1, P2, kout);
      bn_finalize_kernel<<<1, kout, 0, stream>>>(P1, P2, gamma[s][l], beta[s][l],
                                                 scale, shift, nblk, kout, 1.0f / (float)M);
      size_t total = (size_t)M * kout;
      if (l < 2) {
        _Float16* Anext = (Ain == A0) ? A1 : A0;
        bn_relu_h_kernel<<<(unsigned)((total + 255) / 256), 256, 0, stream>>>(
            Y, scale, shift, Anext, total, kout);
        Ain = Anext; kpad = kout; kreal = kout;
      } else {
        bn_relu_f_kernel<<<(unsigned)((total + 255) / 256), 256, 0, stream>>>(
            Y, scale, shift, total, kout);
        maxpool_kernel<<<(BATCH * KCENT * kout + 255) / 256, 256, 0, stream>>>(
            Y, out + BATCH * 3 * KCENT, ns, kout, COFF[s]);
      }
    }
  }
}
